// DomainGenerator_3994319585342
// MI455X (gfx1250) — compile-verified, matrix operand order checked by reading, not run on hardware
//
#include <hip/hip_runtime.h>

// ---------------------------------------------------------------------------
// BiLSTM + combiner LSTM + vocab head for MI455X (gfx1250, wave32, WMMA).
// All GEMMs run through v_wmma_f32_16x16x32_bf16 (bf16 inputs, f32 accum).
// Time recurrence is expressed as a stream-ordered launch sequence (graph-
// capturable, deterministic). Weights are converted f32->bf16 per launch.
//
// Recurrent gate GEMMs (M=256, latency-bound, repeated 64x on the critical
// path) use 4-wave blocks / 64x64 tiles for maximum WGP spread; the big
// vocab-head GEMM uses 8-wave blocks / 64x128 tiles for weight reuse.
// ---------------------------------------------------------------------------

#define BATCH 256
#define SEQ   64
#define HID   512
#define VOCAB 32000

typedef __attribute__((ext_vector_type(16))) __bf16        bf16x16;
typedef __attribute__((ext_vector_type(8)))  float         floatx8;
typedef __attribute__((ext_vector_type(8)))  unsigned int  uintx8;

// ---- helpers ---------------------------------------------------------------

__device__ __forceinline__ unsigned short f2bf(float x) {
  unsigned int u = __float_as_uint(x);
  u += 0x7fffu + ((u >> 16) & 1u);          // round-to-nearest-even
  return (unsigned short)(u >> 16);
}

__device__ __forceinline__ bf16x16 load_bf16x16(const unsigned short* p) {
  // 16 contiguous bf16 = 32 bytes; all call sites are 32B-aligned.
  uintx8 u = *(const uintx8*)p;
  return __builtin_bit_cast(bf16x16, u);
}

__device__ __forceinline__ float sigmoidf(float x) {
  return 1.0f / (1.0f + __expf(-x));
}

// ---- tiny elementwise kernels ---------------------------------------------

__global__ void cvt_f32_bf16(const float* __restrict__ in,
                             unsigned short* __restrict__ out, size_t n) {
  size_t i = (size_t)blockIdx.x * blockDim.x + threadIdx.x;
  size_t stride = (size_t)gridDim.x * blockDim.x;
  for (; i < n; i += stride) out[i] = f2bf(in[i]);
}

__global__ void copy_f32(const float* __restrict__ in,
                         float* __restrict__ out, size_t n) {
  size_t i = (size_t)blockIdx.x * blockDim.x + threadIdx.x;
  size_t stride = (size_t)gridDim.x * blockDim.x;
  for (; i < n; i += stride) out[i] = in[i];
}

// Embedding gather with padding_idx=0, straight into bf16 "seq" buffer.
// seq row r (r = s*B + b) uses token x.flat[r]  (faithful raw reshape).
__global__ void embed_bf16(const int* __restrict__ x,
                           const float* __restrict__ Wemb,
                           unsigned short* __restrict__ out) {
  size_t idx = (size_t)blockIdx.x * blockDim.x + threadIdx.x;
  const size_t total = (size_t)SEQ * BATCH * HID;
  if (idx >= total) return;
  int r = (int)(idx >> 9);            // HID == 512
  int c = (int)(idx & 511);
  int tok = x[r];
  float v = (tok == 0) ? 0.0f : Wemb[(size_t)tok * HID + c];
  out[idx] = f2bf(v);
}

// ---- WMMA GEMM:  out[M=256, N] = A1·W1^T + A2·W2^T (+ bias) --------------
// A* : bf16 row-major (M x K*), W* : bf16 row-major (N x K*)  (i.e. x @ W.T)
// Wave tile 32x32 (4 WMMA accumulators). Blocks may be 4 waves (64x64 tile,
// occupancy-optimized) or 8 waves (64x128 tile, reuse-optimized):
//   waves laid out 2 along M x (nwaves/2) along N.
// grid = (N / (32*(nwaves/2)), 256/64).

__device__ __forceinline__ floatx8 wmma_bf16(bf16x16 a, bf16x16 b, floatx8 c) {
  return __builtin_amdgcn_wmma_f32_16x16x32_bf16(
      /*neg_a=*/false, a, /*neg_b=*/false, b,
      /*c_mod=*/(short)0, c, /*reuse_a=*/false, /*reuse_b=*/false);
}

__device__ __forceinline__ void gemm_accum(const unsigned short* __restrict__ A,
                                           const unsigned short* __restrict__ W,
                                           int K, int m0, int n0, int lane,
                                           floatx8 acc[2][2]) {
  const int rlane = lane & 15;
  const int rsel16 = (lane >> 4) << 4;   // 0 or 16
  const unsigned short* a0p = A + (size_t)(m0 + rlane) * K + rsel16;
  const unsigned short* a1p = A + (size_t)(m0 + 16 + rlane) * K + rsel16;
  const unsigned short* b0p = W + (size_t)(n0 + rlane) * K + rsel16;
  const unsigned short* b1p = W + (size_t)(n0 + 16 + rlane) * K + rsel16;
#pragma unroll 2
  for (int kb = 0; kb < K; kb += 32) {
    // prefetch next weight K-slab (emits global_prefetch_b8 on gfx1250)
    if (kb + 32 < K) {
      __builtin_prefetch((const void*)(b0p + 32), 0, 1);
      __builtin_prefetch((const void*)(b1p + 32), 0, 1);
    }
    bf16x16 a0 = load_bf16x16(a0p);
    bf16x16 a1 = load_bf16x16(a1p);
    bf16x16 b0 = load_bf16x16(b0p);
    bf16x16 b1 = load_bf16x16(b1p);
    acc[0][0] = wmma_bf16(a0, b0, acc[0][0]);
    acc[0][1] = wmma_bf16(a0, b1, acc[0][1]);
    acc[1][0] = wmma_bf16(a1, b0, acc[1][0]);
    acc[1][1] = wmma_bf16(a1, b1, acc[1][1]);
    a0p += 32; a1p += 32; b0p += 32; b1p += 32;
  }
}

__global__ void wmma_gemm_dual(const unsigned short* __restrict__ A1,
                               const unsigned short* __restrict__ W1, int K1,
                               const unsigned short* __restrict__ A2,
                               const unsigned short* __restrict__ W2, int K2,
                               const float* __restrict__ bias,
                               float* __restrict__ out, int N) {
  const int lane   = threadIdx.x & 31;
  const int wave   = threadIdx.x >> 5;
  const int nwaves = blockDim.x >> 5;      // 4 or 8
  const int wm = wave & 1;                 // 2 waves along M
  const int wn = wave >> 1;                // nwaves/2 waves along N
  const int bn = (nwaves >> 1) << 5;       // block N-width: 64 or 128
  const int m0 = blockIdx.y * 64 + wm * 32;
  const int n0 = blockIdx.x * bn + wn * 32;

  floatx8 zero = {0.f, 0.f, 0.f, 0.f, 0.f, 0.f, 0.f, 0.f};
  floatx8 acc[2][2] = {{zero, zero}, {zero, zero}};

  gemm_accum(A1, W1, K1, m0, n0, lane, acc);
  if (A2 != nullptr && K2 > 0) gemm_accum(A2, W2, K2, m0, n0, lane, acc);

  // C/D layout: VGPR v -> M = v (+8 for lanes 16..31); N = lane & 15
  const int mlo = ((lane >> 4) << 3);
  const int nlo = (lane & 15);
#pragma unroll
  for (int mi = 0; mi < 2; ++mi)
#pragma unroll
    for (int ni = 0; ni < 2; ++ni) {
      const int n = n0 + ni * 16 + nlo;
      const float b = bias ? bias[n] : 0.0f;
#pragma unroll
      for (int v = 0; v < 8; ++v) {
        const int m = m0 + mi * 16 + mlo + v;
        out[(size_t)m * N + n] = acc[mi][ni][v] + b;
      }
    }
}

// ---- LSTM pointwise --------------------------------------------------------
// G: raw gates (B x 4*hid) in PyTorch i,f,g,o order (no bias yet).
// Updates c (f32), writes h (bf16) for the next recurrent GEMM, and
// optionally scatters h (bf16) into the combiner input buffer.

__global__ void lstm_pointwise(const float* __restrict__ G,
                               const float* __restrict__ bih,
                               const float* __restrict__ bhh,
                               float* __restrict__ c,
                               unsigned short* __restrict__ h_bf,
                               unsigned short* __restrict__ comb,
                               int comb_stride, int hid) {
  int idx = blockIdx.x * blockDim.x + threadIdx.x;
  int total = BATCH * hid;
  if (idx >= total) return;
  int b = idx / hid;
  int j = idx - b * hid;
  const float* g = G + (size_t)b * 4 * hid;

  float gi = g[j]           + bih[j]           + bhh[j];
  float gf = g[hid + j]     + bih[hid + j]     + bhh[hid + j];
  float gg = g[2 * hid + j] + bih[2 * hid + j] + bhh[2 * hid + j];
  float go = g[3 * hid + j] + bih[3 * hid + j] + bhh[3 * hid + j];

  float i_ = sigmoidf(gi);
  float f_ = sigmoidf(gf);
  float t_ = tanhf(gg);
  float o_ = sigmoidf(go);

  float c2 = f_ * c[idx] + i_ * t_;
  c[idx] = c2;
  float h2 = o_ * tanhf(c2);
  unsigned short hb = f2bf(h2);
  h_bf[idx] = hb;
  if (comb) comb[(size_t)b * comb_stride + j] = hb;
}

// ---- host side -------------------------------------------------------------

extern "C" void kernel_launch(void* const* d_in, const int* in_sizes, int n_in,
                              void* d_out, int out_size, void* d_ws, size_t ws_size,
                              hipStream_t stream) {
  (void)in_sizes; (void)n_in; (void)out_size; (void)ws_size;

  const int*   x      = (const int*)  d_in[0];
  const float* W_emb  = (const float*)d_in[1];
  const float* Wih_f  = (const float*)d_in[2];
  const float* Whh_f  = (const float*)d_in[3];
  const float* bih_f  = (const float*)d_in[4];
  const float* bhh_f  = (const float*)d_in[5];
  const float* Wih_b  = (const float*)d_in[6];
  const float* Whh_b  = (const float*)d_in[7];
  const float* bih_b  = (const float*)d_in[8];
  const float* bhh_b  = (const float*)d_in[9];
  const float* Wih_c  = (const float*)d_in[10];
  const float* Whh_c  = (const float*)d_in[11];
  const float* bih_c  = (const float*)d_in[12];
  const float* bhh_c  = (const float*)d_in[13];
  const float* Wout   = (const float*)d_in[14];
  const float* bout   = (const float*)d_in[15];
  const float* h0f    = (const float*)d_in[16];
  const float* c0f    = (const float*)d_in[17];
  const float* h0b    = (const float*)d_in[18];
  const float* c0b    = (const float*)d_in[19];
  const float* h0c    = (const float*)d_in[20];
  const float* c0c    = (const float*)d_in[21];
  float* out = (float*)d_out;

  // ---- workspace carve-up (all 256B aligned; total ~153 MB) ----
  char* ws = (char*)d_ws;
  size_t off = 0;
  auto alloc = [&](size_t bytes) -> char* {
    char* p = ws + off;
    off += (bytes + 255) & ~(size_t)255;
    return p;
  };
  unsigned short* seq_bf   = (unsigned short*)alloc((size_t)SEQ * BATCH * HID * 2);
  unsigned short* wihf_bf  = (unsigned short*)alloc((size_t)4 * HID * HID * 2);
  unsigned short* whhf_bf  = (unsigned short*)alloc((size_t)4 * HID * HID * 2);
  unsigned short* wihb_bf  = (unsigned short*)alloc((size_t)4 * HID * HID * 2);
  unsigned short* whhb_bf  = (unsigned short*)alloc((size_t)4 * HID * HID * 2);
  unsigned short* wihc_bf  = (unsigned short*)alloc((size_t)8 * HID * 2 * HID * 2);
  unsigned short* whhc_bf  = (unsigned short*)alloc((size_t)8 * HID * 2 * HID * 2);
  unsigned short* wout_bf  = (unsigned short*)alloc((size_t)VOCAB * 2 * HID * 2);
  unsigned short* comb_all = (unsigned short*)alloc((size_t)SEQ * BATCH * 2 * HID * 2);
  float*          Gf       = (float*)alloc((size_t)BATCH * 4 * HID * 4);
  float*          Gb       = (float*)alloc((size_t)BATCH * 4 * HID * 4);
  float*          Gc       = (float*)alloc((size_t)BATCH * 8 * HID * 4);
  unsigned short* h_f      = (unsigned short*)alloc((size_t)BATCH * HID * 2);
  unsigned short* h_b      = (unsigned short*)alloc((size_t)BATCH * HID * 2);
  unsigned short* h_c      = (unsigned short*)alloc((size_t)BATCH * 2 * HID * 2);
  float*          c_f      = (float*)alloc((size_t)BATCH * HID * 4);
  float*          c_b      = (float*)alloc((size_t)BATCH * HID * 4);
  float*          c_c      = (float*)alloc((size_t)BATCH * 2 * HID * 4);

  const int T = 256;
  auto cvt = [&](const float* src, unsigned short* dst, size_t n) {
    int blocks = (int)((n + T - 1) / T);
    if (blocks > 4096) blocks = 4096;
    hipLaunchKernelGGL(cvt_f32_bf16, dim3(blocks), dim3(T), 0, stream, src, dst, n);
  };
  auto cpy = [&](const float* src, float* dst, size_t n) {
    int blocks = (int)((n + T - 1) / T);
    hipLaunchKernelGGL(copy_f32, dim3(blocks), dim3(T), 0, stream, src, dst, n);
  };

  // Phase 0: embedding gather + weight/state conversions
  {
    size_t total = (size_t)SEQ * BATCH * HID;
    hipLaunchKernelGGL(embed_bf16, dim3((unsigned)((total + T - 1) / T)), dim3(T),
                       0, stream, x, W_emb, seq_bf);
  }
  cvt(Wih_f, wihf_bf, (size_t)4 * HID * HID);
  cvt(Whh_f, whhf_bf, (size_t)4 * HID * HID);
  cvt(Wih_b, wihb_bf, (size_t)4 * HID * HID);
  cvt(Whh_b, whhb_bf, (size_t)4 * HID * HID);
  cvt(Wih_c, wihc_bf, (size_t)8 * HID * 2 * HID);
  cvt(Whh_c, whhc_bf, (size_t)8 * HID * 2 * HID);
  cvt(Wout,  wout_bf, (size_t)VOCAB * 2 * HID);
  cvt(h0f, h_f, (size_t)BATCH * HID);
  cvt(h0b, h_b, (size_t)BATCH * HID);
  cvt(h0c, h_c, (size_t)BATCH * 2 * HID);
  cpy(c0f, c_f, (size_t)BATCH * HID);
  cpy(c0b, c_b, (size_t)BATCH * HID);
  cpy(c0c, c_c, (size_t)BATCH * 2 * HID);

  // Phase 1: forward + backward cells (H=512, gates N=2048, K=512+512)
  // 4-wave blocks, 64x64 tiles -> 128 workgroups per GEMM (occupancy).
  const dim3 gemm_grid_fb(4 * HID / 64, BATCH / 64);      // (32, 4)
  const int  pw_blocks_fb = (BATCH * HID) / T;            // 512
  for (int t = 0; t < SEQ; ++t) {
    const unsigned short* xf = seq_bf + (size_t)t * BATCH * HID;
    const unsigned short* xb = seq_bf + (size_t)(SEQ - 1 - t) * BATCH * HID;
    // forward cell
    hipLaunchKernelGGL(wmma_gemm_dual, gemm_grid_fb, dim3(128), 0, stream,
                       xf, wihf_bf, HID, h_f, whhf_bf, HID,
                       (const float*)nullptr, Gf, 4 * HID);
    hipLaunchKernelGGL(lstm_pointwise, dim3(pw_blocks_fb), dim3(T), 0, stream,
                       Gf, bih_f, bhh_f, c_f, h_f,
                       comb_all + (size_t)t * BATCH * 2 * HID, 2 * HID, HID);
    // backward cell (consumes reversed seq, writes produced-order slot t)
    hipLaunchKernelGGL(wmma_gemm_dual, gemm_grid_fb, dim3(128), 0, stream,
                       xb, wihb_bf, HID, h_b, whhb_bf, HID,
                       (const float*)nullptr, Gb, 4 * HID);
    hipLaunchKernelGGL(lstm_pointwise, dim3(pw_blocks_fb), dim3(T), 0, stream,
                       Gb, bih_b, bhh_b, c_b, h_b,
                       comb_all + (size_t)t * BATCH * 2 * HID + HID, 2 * HID, HID);
  }

  // Phase 2: combiner cell (H=1024, gates N=4096, K=1024+1024)
  // 4-wave blocks, 64x64 tiles -> 256 workgroups per GEMM.
  const dim3 gemm_grid_c(8 * HID / 64, BATCH / 64);       // (64, 4)
  const int  pw_blocks_c = (BATCH * 2 * HID) / T;         // 1024
  for (int t = 0; t < SEQ; ++t) {
    const unsigned short* xc = comb_all + (size_t)t * BATCH * 2 * HID;
    hipLaunchKernelGGL(wmma_gemm_dual, gemm_grid_c, dim3(128), 0, stream,
                       xc, wihc_bf, 2 * HID, h_c, whhc_bf, 2 * HID,
                       (const float*)nullptr, Gc, 8 * HID);
    hipLaunchKernelGGL(lstm_pointwise, dim3(pw_blocks_c), dim3(T), 0, stream,
                       Gc, bih_c, bhh_c, c_c, h_c,
                       (unsigned short*)nullptr, 0, 2 * HID);
  }

  // Phase 3: vocab head  out = hc[-1] @ Wout^T + bout   (M=256,N=32000,K=1024)
  // 8-wave blocks, 64x128 tiles -> 1000 workgroups (weight-reuse optimized).
  const dim3 gemm_grid_head(VOCAB / 128, BATCH / 64);     // (250, 4)
  hipLaunchKernelGGL(wmma_gemm_dual, gemm_grid_head, dim3(T), 0, stream,
                     h_c, wout_bf, 2 * HID,
                     (const unsigned short*)nullptr, (const unsigned short*)nullptr, 0,
                     bout, out, VOCAB);
}
